// Henrion_MPNN_67388036874513
// MI455X (gfx1250) — matrix operand order repaired and checked by reading, not compiled, not run on hardware
//
#include <hip/hip_runtime.h>
#include <hip/hip_bf16.h>
#include <stdint.h>

#define N_NODES 65536
#define N_EDGES 1048576
#define NGRAPH  256
// DIN=32, D=64, GRU in = 96, GRU gates = 192

typedef __attribute__((ext_vector_type(16))) __bf16 bf16x16;
typedef __attribute__((ext_vector_type(8)))  float  f32x8;

struct U4 { unsigned x, y, z, w; };

__device__ __forceinline__ unsigned short f2bf(float f) {
  unsigned u = __builtin_bit_cast(unsigned, f);
  u += 0x7FFFu + ((u >> 16) & 1u);          // round-to-nearest-even
  return (unsigned short)(u >> 16);
}
__device__ __forceinline__ float sigm(float x) { return 1.f / (1.f + __expf(-x)); }

// ---------------- converts / fills ----------------
__global__ void k_cvt_bf16(const float* __restrict__ s, unsigned short* __restrict__ d, int n) {
  int i = blockIdx.x * blockDim.x + threadIdx.x;
  if (i < n) d[i] = f2bf(s[i]);
}
__global__ void k_fill_f32(float* p, float v, int n) {
  int i = blockIdx.x * blockDim.x + threadIdx.x;
  if (i < n) p[i] = v;
}
__global__ void k_fill_u32(unsigned* p, unsigned v, int n) {
  int i = blockIdx.x * blockDim.x + threadIdx.x;
  if (i < n) p[i] = v;
}
// m (N x 64 f32) -> mx_bf[:, 0:64]
__global__ void k_m_to_mx(const float* __restrict__ m, unsigned short* __restrict__ mx) {
  int t = blockIdx.x * blockDim.x + threadIdx.x;
  int i = t >> 6, c = t & 63;
  mx[(size_t)i * 96 + c] = f2bf(m[t]);
}
// x (N x 32 f32) -> mx_bf[:, 64:96]   (concat([m, x]) order)
__global__ void k_x_to_mx(const float* __restrict__ x, unsigned short* __restrict__ mx) {
  int t = blockIdx.x * blockDim.x + threadIdx.x;
  int i = t >> 5, c = t & 31;
  mx[(size_t)i * 96 + 64 + c] = f2bf(x[t]);
}

// ---------------- WMMA GEMM:  C[M,NOUT] = A_bf[M,K] @ W_bf[NOUT,K]^T + bias ----------------
// One wave computes one 16x16 output tile.  K in {32,64,96}.
// A-frag lane L: row = tm*16 + (L&15), half h = L>>4; elems 0..7  = A[row, kk+8h .. +7]
//                                                     elems 8..15 = A[row, kk+16+8h .. +7]
// B-frag lane L: wrow = tn*16 + (L&15); elems 0..7 = W[wrow, kk+8h .. +7], elems 8..15 = W[wrow, kk+16+8h .. +7]   (B = W^T)
__global__ void k_gemm_bf16(const __bf16* __restrict__ A, const __bf16* __restrict__ W,
                            const float* __restrict__ bias, float* __restrict__ C,
                            int K, int NOUT, int tiles_n) {
  int wid  = blockIdx.x * (blockDim.x >> 5) + (threadIdx.x >> 5);
  int lane = threadIdx.x & 31;
  int tn = wid % tiles_n;
  int tm = wid / tiles_n;
  int r  = lane & 15;
  int hh = lane >> 4;
  const __bf16* arow = A + (size_t)(tm * 16 + r) * K;
  const __bf16* wrow = W + (size_t)(tn * 16 + r) * K;
  f32x8 acc = {};
  for (int kk = 0; kk < K; kk += 32) {
    union { U4 q[2]; bf16x16 v; } fa, fb;
    fa.q[0] = *(const U4*)(arow + kk + 8 * hh);
    fa.q[1] = *(const U4*)(arow + kk + 16 + 8 * hh);
    fb.q[0] = *(const U4*)(wrow + kk + 8 * hh);
    fb.q[1] = *(const U4*)(wrow + kk + 16 + 8 * hh);
    acc = __builtin_amdgcn_wmma_f32_16x16x32_bf16(false, fa.v, false, fb.v,
                                                  (short)0, acc, false, false);
  }
  // C layout: VGPR j, lanes 0-15: (M=j, N=lane); lanes 16-31: (M=8+j, N=lane-16)
  float bv  = bias[tn * 16 + r];
  int   col = tn * 16 + r;
  int  row0 = tm * 16 + 8 * hh;
#pragma unroll
  for (int j = 0; j < 8; ++j)
    C[(size_t)(row0 + j) * NOUT + col] = acc[j] + bv;
}

// ---------------- edge scatter: m[dst] += lin[src]  (wave per edge, 2 ch/lane) ----------------
__global__ void k_scatter(const float* __restrict__ lin, const int* __restrict__ src,
                          const int* __restrict__ dst, float* __restrict__ m) {
  size_t gid = (size_t)blockIdx.x * blockDim.x + threadIdx.x;
  int e = (int)(gid >> 5);
  int l = (int)(gid & 31);
  if (l == 0 && e + 8 < N_EDGES)
    __builtin_prefetch(lin + (size_t)src[e + 8] * 64, 0, 1);   // global_prefetch_b8
  int s = src[e], d = dst[e];
  float2 v = *(const float2*)(lin + (size_t)s * 64 + 2 * l);
  atomicAdd(m + (size_t)d * 64 + 2 * l,     v.x);
  atomicAdd(m + (size_t)d * 64 + 2 * l + 1, v.y);
}

// ---------------- GRU update (gh = h @ W_hh^T fused; h staged in LDS) ----------------
__global__ void k_gru(const float* __restrict__ gi, const float* hin,
                      const float* __restrict__ Whh, const float* __restrict__ bhh,
                      float* hout, unsigned short* __restrict__ hbf) {
  __shared__ float sh[4][64];
  int t = threadIdx.x;
  int ln = t >> 6, c = t & 63;
  size_t i = (size_t)blockIdx.x * 4 + ln;
  sh[ln][c] = hin[i * 64 + c];
  __syncthreads();
  float gh[3];
#pragma unroll
  for (int g = 0; g < 3; ++g) {
    int row = g * 64 + c;
    const float* w = Whh + (size_t)row * 64;
    float s = bhh[row];
#pragma unroll 8
    for (int k = 0; k < 64; ++k) s += sh[ln][k] * w[k];
    gh[g] = s;
  }
  const float* gr = gi + i * 192;
  float rg = sigm(gr[c] + gh[0]);
  float zg = sigm(gr[64 + c] + gh[1]);
  float ng = tanhf(gr[128 + c] + rg * gh[2]);
  float hv = (1.f - zg) * ng + zg * sh[ln][c];
  hout[i * 64 + c] = hv;
  hbf[i * 64 + c]  = f2bf(hv);
}

// ---------------- Set2Set LSTM cell (B=256, tiny; out-of-place hl) ----------------
__global__ void k_lstm(const float* __restrict__ qs, const float* __restrict__ Wih,
                       const float* __restrict__ bih, const float* __restrict__ Whh,
                       const float* __restrict__ bhh, const float* __restrict__ hin,
                       float* __restrict__ hout, float* __restrict__ cl) {
  int t = blockIdx.x * blockDim.x + threadIdx.x;   // NGRAPH*64
  int b = t >> 6, c = t & 63;
  const float* q  = qs  + (size_t)b * 128;
  const float* hr = hin + (size_t)b * 64;
  float g[4];
#pragma unroll
  for (int gg = 0; gg < 4; ++gg) {
    int row = gg * 64 + c;
    float s = bih[row] + bhh[row];
    const float* wi = Wih + (size_t)row * 128;
    for (int k = 0; k < 128; ++k) s += q[k] * wi[k];
    const float* wh = Whh + (size_t)row * 64;
    for (int k = 0; k < 64; ++k) s += hr[k] * wh[k];
    g[gg] = s;
  }
  float cv = sigm(g[1]) * cl[t] + sigm(g[0]) * tanhf(g[2]);
  cl[t]   = cv;
  hout[t] = sigm(g[3]) * tanhf(cv);
}

// ---------------- attention: e = <h_i, q_{batch[i]}>, segment max via ordered-uint atomicMax ----
__global__ void k_edot(const float* __restrict__ h, const float* __restrict__ q,
                       const int* __restrict__ batch, float* __restrict__ e,
                       unsigned* __restrict__ emax) {
  int i = blockIdx.x * blockDim.x + threadIdx.x;
  int b = batch[i];
  const float4* h4 = (const float4*)(h + (size_t)i * 64);
  const float4* q4 = (const float4*)(q + (size_t)b * 64);
  float s = 0.f;
#pragma unroll
  for (int k = 0; k < 16; ++k) {
    float4 a = h4[k], w = q4[k];
    s += a.x * w.x + a.y * w.y + a.z * w.z + a.w * w.w;
  }
  e[i] = s;
  unsigned u   = __builtin_bit_cast(unsigned, s);
  unsigned key = (u & 0x80000000u) ? ~u : (u | 0x80000000u);
  atomicMax(emax + b, key);
}
__global__ void k_aexp(const float* __restrict__ e, const int* __restrict__ batch,
                       const unsigned* __restrict__ emax, float* __restrict__ a,
                       float* __restrict__ asum) {
  int i = blockIdx.x * blockDim.x + threadIdx.x;
  int b = batch[i];
  unsigned key = emax[b];
  float em = 0.f;                                  // empty segment -> 0 (isfinite guard)
  if (key != 0u) {
    unsigned u = (key & 0x80000000u) ? (key ^ 0x80000000u) : ~key;
    em = __builtin_bit_cast(float, u);
  }
  float av = expf(e[i] - em);
  a[i] = av;
  atomicAdd(asum + b, av);
}
__global__ void k_racc(const float* __restrict__ h, const float* __restrict__ a,
                       const float* __restrict__ asum, const int* __restrict__ batch,
                       float* __restrict__ r) {
  int t = blockIdx.x * blockDim.x + threadIdx.x;   // N*64
  int i = t >> 6, c = t & 63;
  int b = batch[i];
  float coef = a[i] / fmaxf(asum[b], 1e-16f);
  atomicAdd(r + (size_t)b * 64 + c, coef * h[t]);
}
__global__ void k_qstar(const float* __restrict__ hl, const float* __restrict__ r,
                        float* __restrict__ qs) {
  int t = blockIdx.x * blockDim.x + threadIdx.x;   // NGRAPH*64
  int b = t >> 6, c = t & 63;
  qs[(size_t)b * 128 + c]      = hl[t];
  qs[(size_t)b * 128 + 64 + c] = r[t];
}

// ---------------- head: relu MLP -> logits -> log_softmax ----------------
__global__ void k_head(const float* __restrict__ qs, const float* __restrict__ W1,
                       const float* __restrict__ b1, const float* __restrict__ W2,
                       const float* __restrict__ b2, float* __restrict__ out) {
  int b = blockIdx.x * blockDim.x + threadIdx.x;
  if (b >= NGRAPH) return;
  const float* q = qs + (size_t)b * 128;
  float hid[64];
  for (int j = 0; j < 64; ++j) {
    float s = b1[j];
    const float* w = W1 + (size_t)j * 128;
    for (int k = 0; k < 128; ++k) s += q[k] * w[k];
    hid[j] = fmaxf(s, 0.f);
  }
  float lg[4], mx = -1e30f;
  for (int o = 0; o < 4; ++o) {
    float s = b2[o];
    const float* w = W2 + (size_t)o * 64;
    for (int k = 0; k < 64; ++k) s += hid[k] * w[k];
    lg[o] = s;
    mx = fmaxf(mx, s);
  }
  float se = 0.f;
  for (int o = 0; o < 4; ++o) se += expf(lg[o] - mx);
  float lse = mx + logf(se);
  for (int o = 0; o < 4; ++o) out[b * 4 + o] = lg[o] - lse;
}

extern "C" void kernel_launch(void* const* d_in, const int* in_sizes, int n_in,
                              void* d_out, int out_size, void* d_ws, size_t ws_size,
                              hipStream_t stream) {
  (void)in_sizes; (void)n_in; (void)out_size; (void)ws_size;
  const float* x      = (const float*)d_in[0];
  const int*   ei     = (const int*)d_in[1];
  const int*   src    = ei;
  const int*   dst    = ei + N_EDGES;
  const int*   batch  = (const int*)d_in[2];
  const float* W_mlp  = (const float*)d_in[3];
  const float* b_mlp  = (const float*)d_in[4];
  const float* W_conv = (const float*)d_in[5];
  const float* b_conv = (const float*)d_in[6];
  const float* gWih   = (const float*)d_in[7];
  const float* gWhh   = (const float*)d_in[8];
  const float* gbih   = (const float*)d_in[9];
  const float* gbhh   = (const float*)d_in[10];
  const float* lWih   = (const float*)d_in[11];
  const float* lWhh   = (const float*)d_in[12];
  const float* lbih   = (const float*)d_in[13];
  const float* lbhh   = (const float*)d_in[14];
  const float* W1     = (const float*)d_in[15];
  const float* b1     = (const float*)d_in[16];
  const float* W2     = (const float*)d_in[17];
  const float* b2     = (const float*)d_in[18];
  float* out = (float*)d_out;

  char* base = (char*)d_ws;
  size_t off = 0;
  auto alloc = [&](size_t bytes) { void* p = base + off; off += (bytes + 255) & ~(size_t)255; return p; };
  float*          h     = (float*)alloc((size_t)N_NODES * 64 * 4);
  unsigned short* h_bf  = (unsigned short*)alloc((size_t)N_NODES * 64 * 2);
  float*          lin   = (float*)alloc((size_t)N_NODES * 64 * 4);
  float*          m     = (float*)alloc((size_t)N_NODES * 64 * 4);
  unsigned short* mx_bf = (unsigned short*)alloc((size_t)N_NODES * 96 * 2);
  float*          gi    = (float*)alloc((size_t)N_NODES * 192 * 4);
  unsigned short* x_bf  = (unsigned short*)alloc((size_t)N_NODES * 32 * 2);
  float*          ebuf  = (float*)alloc((size_t)N_NODES * 4);
  float*          abuf  = (float*)alloc((size_t)N_NODES * 4);
  unsigned short* Wmlp_bf  = (unsigned short*)alloc(64 * 32 * 2);
  unsigned short* Wconv_bf = (unsigned short*)alloc(64 * 64 * 2);
  unsigned short* Wih_bf   = (unsigned short*)alloc(192 * 96 * 2);
  float*    qstar = (float*)alloc(NGRAPH * 128 * 4);
  float*    hl0   = (float*)alloc(NGRAPH * 64 * 4);
  float*    hl1   = (float*)alloc(NGRAPH * 64 * 4);
  float*    cl    = (float*)alloc(NGRAPH * 64 * 4);
  unsigned* emax  = (unsigned*)alloc(NGRAPH * 4);
  float*    asum  = (float*)alloc(NGRAPH * 4);
  float*    rbuf  = (float*)alloc(NGRAPH * 64 * 4);

  const int T = 256;
  auto cdiv = [](int a, int b) { return (a + b - 1) / b; };

  // bf16 staging
  k_cvt_bf16<<<cdiv(N_NODES * 32, T), T, 0, stream>>>(x, x_bf, N_NODES * 32);
  k_cvt_bf16<<<cdiv(64 * 32, T),  T, 0, stream>>>(W_mlp,  Wmlp_bf,  64 * 32);
  k_cvt_bf16<<<cdiv(64 * 64, T),  T, 0, stream>>>(W_conv, Wconv_bf, 64 * 64);
  k_cvt_bf16<<<cdiv(192 * 96, T), T, 0, stream>>>(gWih,   Wih_bf,   192 * 96);
  k_x_to_mx<<<cdiv(N_NODES * 32, T), T, 0, stream>>>(x, mx_bf);

  // conv1.mlp: h = x @ W_mlp^T + b   (WMMA, K=32, NOUT=64)
  k_gemm_bf16<<<(N_NODES / 16) * 4 / 8, T, 0, stream>>>(
      (const __bf16*)x_bf, (const __bf16*)Wmlp_bf, b_mlp, h, 32, 64, 4);
  k_cvt_bf16<<<cdiv(N_NODES * 64, T), T, 0, stream>>>(h, h_bf, N_NODES * 64);

  // 2 message-passing steps
  for (int step = 0; step < 2; ++step) {
    k_gemm_bf16<<<(N_NODES / 16) * 4 / 8, T, 0, stream>>>(
        (const __bf16*)h_bf, (const __bf16*)Wconv_bf, b_conv, lin, 64, 64, 4);
    k_fill_f32<<<cdiv(N_NODES * 64, T), T, 0, stream>>>(m, 0.f, N_NODES * 64);
    k_scatter<<<(N_EDGES * 32) / T, T, 0, stream>>>(lin, src, dst, m);
    k_m_to_mx<<<cdiv(N_NODES * 64, T), T, 0, stream>>>(m, mx_bf);
    k_gemm_bf16<<<(N_NODES / 16) * 12 / 8, T, 0, stream>>>(
        (const __bf16*)mx_bf, (const __bf16*)Wih_bf, gbih, gi, 96, 192, 12);
    k_gru<<<N_NODES / 4, T, 0, stream>>>(gi, h, gWhh, gbhh, h, h_bf);
  }

  // Set2Set
  k_fill_f32<<<cdiv(NGRAPH * 128, T), T, 0, stream>>>(qstar, 0.f, NGRAPH * 128);
  k_fill_f32<<<cdiv(NGRAPH * 64, T),  T, 0, stream>>>(hl0,   0.f, NGRAPH * 64);
  k_fill_f32<<<cdiv(NGRAPH * 64, T),  T, 0, stream>>>(cl,    0.f, NGRAPH * 64);
  float* hin = hl0; float* hout = hl1;
  for (int it = 0; it < 3; ++it) {
    k_lstm<<<cdiv(NGRAPH * 64, T), T, 0, stream>>>(qstar, lWih, lbih, lWhh, lbhh, hin, hout, cl);
    k_fill_u32<<<cdiv(NGRAPH, T),      T, 0, stream>>>(emax, 0u,  NGRAPH);
    k_fill_f32<<<cdiv(NGRAPH, T),      T, 0, stream>>>(asum, 0.f, NGRAPH);
    k_fill_f32<<<cdiv(NGRAPH * 64, T), T, 0, stream>>>(rbuf, 0.f, NGRAPH * 64);
    k_edot<<<N_NODES / T, T, 0, stream>>>(h, hout, batch, ebuf, emax);
    k_aexp<<<N_NODES / T, T, 0, stream>>>(ebuf, batch, emax, abuf, asum);
    k_racc<<<(N_NODES * 64) / T, T, 0, stream>>>(h, abuf, asum, batch, rbuf);
    k_qstar<<<cdiv(NGRAPH * 64, T), T, 0, stream>>>(hout, rbuf, qstar);
    float* tmp = hin; hin = hout; hout = tmp;
  }

  k_head<<<1, 256, 0, stream>>>(qstar, W1, b1, W2, b2, out);
}
